// CorrelationalDetector_25194278158653
// MI455X (gfx1250) — compile-verified, matrix operand order checked by reading, not run on hardware
//
#include <hip/hip_runtime.h>
#include <cstdint>

// ---------------------------------------------------------------------------
// CDNA5 (gfx1250): NHWC implicit-GEMM conv encoder on v_wmma_f32_16x16x32_bf16
// (4 cout-tiles per wave; B fragments = two contiguous b128 loads per lane),
// bf16 activations end-to-end, TDM (tensor_load_to_lds) staging + broadcast
// ds_load_b128 reads for the cross-correlation, single-block BatchNorm.
// ---------------------------------------------------------------------------

typedef __attribute__((ext_vector_type(16))) __bf16 v16bf;
typedef __attribute__((ext_vector_type(8)))  __bf16 v8bf;
typedef __attribute__((ext_vector_type(8)))  float  v8f;
typedef __attribute__((ext_vector_type(4)))  unsigned v4u;
typedef __attribute__((ext_vector_type(8)))  int    v8i;
typedef __attribute__((ext_vector_type(4)))  int    v4i;

// K index inside a 32-wide chunk for 16-bit A/B fragments
// (CDNA5 ISA "16-bit A-Matrix 16x32"):
//   element e -> K = kbase(e) + 8*khalf, kbase = {0..7, 16..23}.
// With NHWC (channel-contiguous K) a fragment is two contiguous 8-element runs.
__device__ __forceinline__ int kmap(int e, int khalf) {
    int v = e >> 1;
    return ((v >> 2) << 4) + (khalf << 3) + ((v & 3) << 1) + (e & 1);
}

__device__ __forceinline__ v16bf frag_from_runs(v8bf lo, v8bf hi) {
    return __builtin_shufflevector(lo, hi, 0, 1, 2, 3, 4, 5, 6, 7,
                                           8, 9, 10, 11, 12, 13, 14, 15);
}

// ---------------------------------------------------------------------------
// Weight pre-swizzle: W[Cout][Cin][3][3] f32 -> bf16 A-fragments
// [coutTile][kChunk][lane][16] with K ordered k = tap*Cinp + cin
// (Cinp = power-of-two padded Cin), zero-padded.
// ---------------------------------------------------------------------------
__global__ void prep_weights_bf16(const float* __restrict__ W,
                                  __bf16* __restrict__ Wb,
                                  int Cin, int lc /*log2 Cinp*/,
                                  int nChunks, int total) {
    int t = blockIdx.x * blockDim.x + threadIdx.x;
    if (t >= total) return;
    int e     = t & 15;
    int lane  = (t >> 4) & 31;
    int tile  = t >> 9;                // coutTile * nChunks + kChunk
    int kc    = tile % nChunks;
    int ct    = tile / nChunks;
    int m     = lane & 15;
    int khalf = lane >> 4;
    int k     = kc * 32 + kmap(e, khalf);
    int tap   = k >> lc;
    int cin   = k & ((1 << lc) - 1);
    int cout  = ct * 16 + m;
    float v = 0.0f;
    if (tap < 9 && cin < Cin)
        v = W[((size_t)cout * Cin + cin) * 9 + tap];
    Wb[(((size_t)ct * nChunks + kc) * 32 + lane) * 16 + e] = (__bf16)v;
}

// ---------------------------------------------------------------------------
// L0 direct conv (Cin=3), f32 NCHW -> f32 NCHW, bias+ReLU.
// ---------------------------------------------------------------------------
__global__ void conv3x3_direct_relu(const float* __restrict__ in,
                                    const float* __restrict__ W,
                                    const float* __restrict__ bias,
                                    float* __restrict__ out,
                                    int Cin, int Hin, int Win,
                                    int Cout, int Hout, int Wout,
                                    int stride, int B) {
    size_t idx   = (size_t)blockIdx.x * blockDim.x + threadIdx.x;
    size_t total = (size_t)B * Cout * Hout * Wout;
    if (idx >= total) return;
    int ow = (int)(idx % Wout); size_t t = idx / Wout;
    int oh = (int)(t % Hout);   t /= Hout;
    int co = (int)(t % Cout);
    int b  = (int)(t / Cout);
    float acc = bias[co];
    for (int ci = 0; ci < Cin; ++ci) {
        const float* ip = in + ((size_t)b * Cin + ci) * Hin * Win;
        const float* wp = W + ((size_t)co * Cin + ci) * 9;
        #pragma unroll
        for (int ky = 0; ky < 3; ++ky) {
            int ih = oh * stride + ky - 1;
            if ((unsigned)ih >= (unsigned)Hin) continue;
            #pragma unroll
            for (int kx = 0; kx < 3; ++kx) {
                int iw = ow * stride + kx - 1;
                if ((unsigned)iw >= (unsigned)Win) continue;
                acc += ip[(size_t)ih * Win + iw] * wp[ky * 3 + kx];
            }
        }
    }
    out[idx] = fmaxf(acc, 0.0f);
}

// ---------------------------------------------------------------------------
// L1 direct conv (Cin=3, Cout=16): f32 NCHW in -> bf16 NHWC out, zero-padded
// to Cpad=32 channels. Channel-fastest thread mapping => coalesced stores.
// ---------------------------------------------------------------------------
__global__ void conv3x3_direct_pad_nhwc(const float* __restrict__ in,
                                        const float* __restrict__ W,
                                        const float* __restrict__ bias,
                                        __bf16* __restrict__ out,
                                        int Cin, int Hin, int Win,
                                        int Cout, int Cpad, int Hout, int Wout,
                                        int stride, int B) {
    size_t idx   = (size_t)blockIdx.x * blockDim.x + threadIdx.x;
    size_t total = (size_t)B * Cpad * Hout * Wout;
    if (idx >= total) return;
    int co = (int)(idx % Cpad); size_t t = idx / Cpad;
    int ow = (int)(t % Wout);   t /= Wout;
    int oh = (int)(t % Hout);
    int b  = (int)(t / Hout);
    if (co >= Cout) { out[idx] = (__bf16)0.0f; return; }   // zero pad channels
    float acc = bias[co];
    for (int ci = 0; ci < Cin; ++ci) {
        const float* ip = in + ((size_t)b * Cin + ci) * Hin * Win;
        const float* wp = W + ((size_t)co * Cin + ci) * 9;
        #pragma unroll
        for (int ky = 0; ky < 3; ++ky) {
            int ih = oh * stride + ky - 1;
            if ((unsigned)ih >= (unsigned)Hin) continue;
            #pragma unroll
            for (int kx = 0; kx < 3; ++kx) {
                int iw = ow * stride + kx - 1;
                if ((unsigned)iw >= (unsigned)Win) continue;
                acc += ip[(size_t)ih * Win + iw] * wp[ky * 3 + kx];
            }
        }
    }
    out[idx] = (__bf16)fmaxf(acc, 0.0f);
}

// ---------------------------------------------------------------------------
// NHWC implicit-GEMM conv via WMMA bf16.
//  - wave tile: 64 couts (4 x 16 register-blocked) x 16 output pixels
//  - K = 9*Cinp tap-major; NHWC => B fragment = 2 contiguous b128 loads/lane
//  - boundary taps redirected to a zeroed pad buffer (no exec juggling)
//  - bias(+ReLU) fused; NHWC bf16 output, b128 stores
// All pre-WMMA branches are wave-uniform (EXEC all-1s at every v_wmma).
// ---------------------------------------------------------------------------
template<int CC>   // Cinp = CC*32 input channels
__global__ void conv3x3_wmma_nhwc(const __bf16* __restrict__ in,   // [B,Hin,Win,CC*32]
                                  const __bf16* __restrict__ Wb,
                                  const float* __restrict__ bias,
                                  __bf16* __restrict__ out,        // [B,Hout,Wout,Cout]
                                  const __bf16* __restrict__ zeroPad,
                                  int Hin, int Win,
                                  int Cout, int Hout, int Wout,
                                  int stride, int relu) {
    constexpr int Cinp    = CC * 32;
    constexpr int nChunks = 9 * CC;
    const int HW   = Hout * Wout;
    const int lane = threadIdx.x & 31;
    const int wave = threadIdx.x >> 5;
    const int pixBase = (blockIdx.x * (blockDim.x >> 5) + wave) * 16;
    if (pixBase >= HW) return;                 // wave-uniform exit, no barriers
    const int ct0 = blockIdx.y * 4;            // first of 4 cout tiles
    const int b   = blockIdx.z;

    const int n     = lane & 15;               // pixel column (N), also M for A
    const int khalf = lane >> 4;
    const int p     = pixBase + n;
    const int pv    = (p < HW);
    const int oh    = pv ? (p / Wout) : 0;
    const int ow    = pv ? (p % Wout) : 0;

    const __bf16* inB = in + (size_t)b * Hin * Win * Cinp;
    const __bf16* Wt  = Wb + (size_t)ct0 * nChunks * 512;
    const size_t tileStride = (size_t)nChunks * 512;

    v8f acc[4] = {};
    for (int tap = 0; tap < 9; ++tap) {
        int ky = tap / 3, kx = tap - (tap / 3) * 3;
        int ih = oh * stride + ky - 1;
        int iw = ow * stride + kx - 1;
        bool ok = pv && (unsigned)ih < (unsigned)Hin && (unsigned)iw < (unsigned)Win;
        const __bf16* s0 = ok ? inB + ((size_t)ih * Win + iw) * Cinp + 8 * khalf
                              : zeroPad;       // zeroPad spans cc*32+24 < 256
        #pragma unroll
        for (int cc = 0; cc < CC; ++cc) {
            const __bf16* sc = s0 + cc * 32;
            v8bf lo = *(const v8bf*)sc;        // K offsets 0..7
            v8bf hi = *(const v8bf*)(sc + 16); // K offsets 16..23
            v16bf bfrag = frag_from_runs(lo, hi);
            const int kc = tap * CC + cc;
            const __bf16* ap = Wt + ((size_t)kc * 32 + lane) * 16;
            #pragma unroll
            for (int t = 0; t < 4; ++t) {
                v16bf af = *(const v16bf*)(ap + (size_t)t * tileStride);
                acc[t] = __builtin_amdgcn_wmma_f32_16x16x32_bf16(
                             false, af, false, bfrag, (short)0, acc[t], false, false);
            }
        }
    }

    // D layout: lane column n (pixel), rows m = 8*khalf + r -> 8 consecutive
    // couts per tile => one b128 NHWC store per tile.
    if (pv) {
        __bf16* op = out + ((size_t)b * HW + p) * Cout + khalf * 8;
        #pragma unroll
        for (int t = 0; t < 4; ++t) {
            v8bf ov;
            #pragma unroll
            for (int r = 0; r < 8; ++r) {
                float v = acc[t][r] + bias[(ct0 + t) * 16 + khalf * 8 + r];
                if (relu) v = fmaxf(v, 0.0f);
                ov[r] = (__bf16)v;
            }
            *(v8bf*)(op + (size_t)(ct0 + t) * 16) = ov;
        }
    }
}

// ---------------------------------------------------------------------------
// Cross-correlation via WMMA + TDM, NHWC feature maps.
// K = 8*8*256 = 16384 ordered (ky,kx,c) == natural NHWC order of crop_fm.
// The 32KB crop vector is DMA'd to LDS once per block by the Tensor Data
// Mover. ALL lanes broadcast-load the same crop runs from LDS (two
// ds_load_b128, conflict-free): B's 16 columns are identical, and since only
// column 0 of D is ever stored, the duplicate columns are simply dead.
// A = 16 frame patches x 32 K, two b128 global loads per lane with
// instruction-immediate channel offsets (pix-outer / cc-unrolled loop).
// Grid sized so every wave reaches the barrier (max posBase = 624 < 625).
// ---------------------------------------------------------------------------
__global__ void xcorr_wmma_tdm(const __bf16* __restrict__ frame_fm, // [64,32,32,256]
                               const __bf16* __restrict__ crop_bf,  // [64,8,8,256]
                               float* __restrict__ rmap) {          // [64,625]
    __shared__ __align__(16) __bf16 scrop[16384];   // 32 KB
    const int lane = threadIdx.x & 31;
    const int wave = threadIdx.x >> 5;
    const int b = blockIdx.z;

    if (wave == 0) {
        // Tensor DMA descriptor (CDNA5 ISA):
        // 1-D tile: 16384 x 2-byte elements, global -> LDS.
        uint64_t ga  = (uint64_t)(uintptr_t)(crop_bf + (size_t)b * 16384);
        unsigned lds = (unsigned)(uintptr_t)&scrop[0];   // low 32 bits = LDS offset
        v4u g0;
        g0[0] = 1u;                                      // count=1
        g0[1] = lds;                                     // lds_addr
        g0[2] = (unsigned)(ga & 0xFFFFFFFFu);            // global_addr[31:0]
        g0[3] = (unsigned)((ga >> 32) & 0x01FFFFFFu) | 0x80000000u; // addr hi | type=2
        v8i g1;
        g1[0] = (int)(1u << 16);                         // data_size = 2 bytes
        g1[1] = (int)(16384u << 16);                     // tensor_dim0[15:0]
        g1[2] = (int)(1u << 16);                         // tensor_dim1 = 1
        g1[3] = (int)(16384u << 16);                     // tile_dim0 = 16384
        g1[4] = 0;                                       // tile_dim1/2 = 0 (1-D)
        g1[5] = 16384;                                   // tensor_dim0_stride
        g1[6] = 0;
        g1[7] = 0;
        v4i g2 = {0, 0, 0, 0};
        v4i g3 = {0, 0, 0, 0};
        v8i g4 = {0, 0, 0, 0, 0, 0, 0, 0};               // 6-arg (clang-23) form
        __builtin_amdgcn_tensor_load_to_lds(g0, g1, g2, g3, g4, 0);
        __builtin_amdgcn_s_wait_tensorcnt(0);
    }
    __syncthreads();

    const int posBase = (blockIdx.x * (blockDim.x >> 5) + wave) * 16;
    const int n     = lane & 15;
    const int khalf = lane >> 4;
    const int pos   = posBase + n;             // M index for A fragment
    const int pv    = (pos < 625);
    const int oy    = pv ? pos / 25 : 0;       // clamped: loads stay in range
    const int ox    = pv ? pos % 25 : 0;

    const __bf16* fb = frame_fm + ((size_t)b * 32 * 32) * 256 + 8 * khalf;
    const __bf16* sb0 = scrop + 8 * khalf;

    v8f acc = {};
    for (int pix = 0; pix < 64; ++pix) {       // patch pixel (ky,kx)
        int ky = pix >> 3, kx = pix & 7;
        const __bf16* sa = fb + (((size_t)(oy + ky) * 32) + (ox + kx)) * 256;
        const __bf16* sb = sb0 + pix * 256;
        #pragma unroll
        for (int cc = 0; cc < 8; ++cc) {       // channel chunks: const offsets
            v16bf afrag = frag_from_runs(*(const v8bf*)(sa + cc * 32),
                                         *(const v8bf*)(sa + cc * 32 + 16));
            v16bf bfrag = frag_from_runs(*(const v8bf*)(sb + cc * 32),
                                         *(const v8bf*)(sb + cc * 32 + 16));
            acc = __builtin_amdgcn_wmma_f32_16x16x32_bf16(
                      false, afrag, false, bfrag, (short)0, acc, false, false);
        }
    }
    if (n == 0) {   // column 0 of D: lanes 0 (rows 0-7) and 16 (rows 8-15)
        #pragma unroll
        for (int r = 0; r < 8; ++r) {
            int pp = posBase + khalf * 8 + r;
            if (pp < 625) rmap[(size_t)b * 625 + pp] = acc[r];
        }
    }
}

// ---------------------------------------------------------------------------
// BatchNorm2d(1), training mode, over 64*625 = 40000 values. Single block.
// ---------------------------------------------------------------------------
__global__ void batchnorm_rmap(const float* __restrict__ rmap,
                               const float* __restrict__ gamma,
                               const float* __restrict__ beta,
                               float* __restrict__ out) {
    __shared__ float ssum[256], ssq[256];
    __shared__ float smean, sinv;
    const int N = 40000;
    int tid = threadIdx.x;
    float s = 0.0f, q = 0.0f;
    for (int i = tid; i < N; i += blockDim.x) {
        float v = rmap[i];
        s += v; q += v * v;
    }
    ssum[tid] = s; ssq[tid] = q;
    __syncthreads();
    for (int off = blockDim.x >> 1; off > 0; off >>= 1) {
        if (tid < off) { ssum[tid] += ssum[tid + off]; ssq[tid] += ssq[tid + off]; }
        __syncthreads();
    }
    if (tid == 0) {
        float mean = ssum[0] / (float)N;
        float var  = ssq[0] / (float)N - mean * mean;
        smean = mean;
        sinv  = rsqrtf(var + 1e-5f);
    }
    __syncthreads();
    float g = gamma[0], bt = beta[0];
    for (int i = tid; i < N; i += blockDim.x)
        out[i] = (rmap[i] - smean) * sinv * g + bt;
}

// ---------------------------------------------------------------------------
// Workspace (all chunks 32B aligned):
//   f0 3,145,728 f32 | rmap 40,960 f32 | zeroPad 256 bf16 |
//   bfA 16,777,216 | bfB 8,388,608 | bfC 1,048,576 | bfD 524,288 |
//   cropBF 1,048,576 | Wb2 18,432 | Wb3 73,728 | Wb4 294,912   (bf16)
// ---------------------------------------------------------------------------
extern "C" void kernel_launch(void* const* d_in, const int* in_sizes, int n_in,
                              void* d_out, int out_size, void* d_ws, size_t ws_size,
                              hipStream_t stream) {
    const float* crop  = (const float*)d_in[0];
    const float* frame = (const float*)d_in[1];
    const float* W0 = (const float*)d_in[2];  const float* b0 = (const float*)d_in[3];
    const float* W1 = (const float*)d_in[4];  const float* b1 = (const float*)d_in[5];
    const float* W2 = (const float*)d_in[6];  const float* b2 = (const float*)d_in[7];
    const float* W3 = (const float*)d_in[8];  const float* b3 = (const float*)d_in[9];
    const float* W4 = (const float*)d_in[10]; const float* b4 = (const float*)d_in[11];
    const float* gamma = (const float*)d_in[12];
    const float* beta  = (const float*)d_in[13];
    float* out = (float*)d_out;

    float*  f0   = (float*)d_ws;
    float*  rmap = f0 + 3145728;
    __bf16* zeroPad = (__bf16*)(rmap + 40960);   // 256 elems, zeroed below
    __bf16* bfA  = zeroPad + 256;
    __bf16* bfB  = bfA + 16777216;
    __bf16* bfC  = bfB + 8388608;
    __bf16* bfD  = bfC + 1048576;
    __bf16* cropBF = bfD + 524288;
    __bf16* Wb2  = cropBF + 1048576;   // Cinp=32 : 4 tiles * 9  chunks * 512
    __bf16* Wb3  = Wb2 + 18432;        // Cinp=64 : 8 tiles * 18 chunks * 512
    __bf16* Wb4  = Wb3 + 73728;        // Cinp=128: 16 tiles * 36 chunks * 512

    const int B = 64;
    dim3 blk256(256);

    hipMemsetAsync(zeroPad, 0, 256 * sizeof(__bf16), stream);  // boundary pad

    // Weight pre-swizzle (tap-major K, Cin padded to power of two)
    {
        int t2 = 4 * 9 * 512, t3 = 8 * 18 * 512, t4 = 16 * 36 * 512;
        prep_weights_bf16<<<dim3((t2 + 255) / 256), blk256, 0, stream>>>(W2, Wb2, 16, 5, 9,  t2);
        prep_weights_bf16<<<dim3((t3 + 255) / 256), blk256, 0, stream>>>(W3, Wb3, 64, 6, 18, t3);
        prep_weights_bf16<<<dim3((t4 + 255) / 256), blk256, 0, stream>>>(W4, Wb4, 128, 7, 36, t4);
    }

    // ---- frame path: 256 -> 128 -> 64 -> 64 -> 32 -> 32 ----
    {
        size_t n0 = (size_t)B * 3 * 128 * 128;
        conv3x3_direct_relu<<<dim3((n0 + 255) / 256), blk256, 0, stream>>>(
            frame, W0, b0, f0, 3, 256, 256, 3, 128, 128, 2, B);
        size_t n1 = (size_t)B * 32 * 64 * 64;
        conv3x3_direct_pad_nhwc<<<dim3((n1 + 255) / 256), blk256, 0, stream>>>(
            f0, W1, b1, bfB, 3, 128, 128, 16, 32, 64, 64, 2, B);
        conv3x3_wmma_nhwc<1><<<dim3(32, 1, B), blk256, 0, stream>>>(   // 4096 px / 128
            bfB, Wb2, b2, bfA, zeroPad, 64, 64, 64, 64, 64, 1, 1);
        conv3x3_wmma_nhwc<2><<<dim3(8, 2, B), blk256, 0, stream>>>(    // 1024 px / 128
            bfA, Wb3, b3, bfB, zeroPad, 64, 64, 128, 32, 32, 2, 1);
        conv3x3_wmma_nhwc<4><<<dim3(8, 4, B), blk256, 0, stream>>>(    // final: no ReLU
            bfB, Wb4, b4, bfA, zeroPad, 32, 32, 256, 32, 32, 1, 0);
        // frame_fm = bfA [64,32,32,256] bf16 NHWC
    }

    // ---- crop path: 64 -> 32 -> 16 -> 16 -> 8 -> 8 ----
    {
        size_t n0 = (size_t)B * 3 * 32 * 32;
        conv3x3_direct_relu<<<dim3((n0 + 255) / 256), blk256, 0, stream>>>(
            crop, W0, b0, f0, 3, 64, 64, 3, 32, 32, 2, B);
        size_t n1 = (size_t)B * 32 * 16 * 16;
        conv3x3_direct_pad_nhwc<<<dim3((n1 + 255) / 256), blk256, 0, stream>>>(
            f0, W1, b1, bfD, 3, 32, 32, 16, 32, 16, 16, 2, B);
        conv3x3_wmma_nhwc<1><<<dim3(2, 1, B), blk256, 0, stream>>>(    // 256 px
            bfD, Wb2, b2, bfC, zeroPad, 16, 16, 64, 16, 16, 1, 1);
        conv3x3_wmma_nhwc<2><<<dim3(1, 2, B), blk256, 0, stream>>>(    // 64 px
            bfC, Wb3, b3, bfD, zeroPad, 16, 16, 128, 8, 8, 2, 1);
        conv3x3_wmma_nhwc<4><<<dim3(1, 4, B), blk256, 0, stream>>>(    // final: no ReLU
            bfD, Wb4, b4, cropBF, zeroPad, 8, 8, 256, 8, 8, 1, 0);
        // crop_fm = cropBF [64,8,8,256] bf16 NHWC (== xcorr K order)
    }

    // cross-correlation: TDM-staged crop in LDS, 40 position tiles / sample
    xcorr_wmma_tdm<<<dim3(5, 1, B), blk256, 0, stream>>>(bfA, cropBF, rmap);

    // BatchNorm over [64,1,25,25]
    batchnorm_rmap<<<dim3(1), blk256, 0, stream>>>(rmap, gamma, beta, out);
}
